// Attention_63161789055023
// MI455X (gfx1250) — matrix operand order repaired and checked by reading, not compiled, not run on hardware
//
#include <hip/hip_runtime.h>
#include <hip/hip_bf16.h>
#include <math.h>

// ---------------------------------------------------------------------------
// Problem constants
// ---------------------------------------------------------------------------
#define BB     16
#define NN     576
#define DIMK   512
#define HEADS  8
#define DH     64
#define INNER  512
#define MROWS  (BB * NN)        // 9216
#define PPOS   (NN * NN)        // 331776
#define SCALE_F 8.0f
#define LEAK    0.1f

typedef __attribute__((ext_vector_type(16))) _Float16 v16h;
typedef __attribute__((ext_vector_type(8)))  float    v8f;
typedef __attribute__((address_space(3))) _Float16 lds_f16_t;

#define DEV __device__ __forceinline__

union Frag16 { uint4 u4[2]; v16h h; };

// A fragment: 16x32 f16 tile, row-major source with leading dim lda.
// lane r(0..15) -> row r; halves 0..7 = K[kb..kb+7], halves 8..15 = K[kb+16..kb+23],
// kb = 8*(lane>>4).
DEV v16h load_a_frag(const _Float16* base, int lda, int lane) {
    int r  = lane & 15;
    int kb = (lane >> 4) << 3;
    const _Float16* pp = base + r * lda + kb;
    Frag16 f;
    f.u4[0] = *(const uint4*)(pp);
    f.u4[1] = *(const uint4*)(pp + 16);
    return f.h;
}

// B fragment from TRANSPOSED layout Bt[Ncol][K] (row-major, ld = ldb).
// lane c(0..15) -> col c; halves 0..7 = Bt[col][kb..kb+7], halves 8..15 = Bt[col][kb+16..kb+23],
// kb = 8*(lane>>4).
DEV v16h load_bt_frag(const _Float16* base, int ldb, int lane) {
    int c  = lane & 15;
    int kb = (lane >> 4) << 3;
    const _Float16* pp = base + c * ldb + kb;
    Frag16 f;
    f.u4[0] = *(const uint4*)(pp);
    f.u4[1] = *(const uint4*)(pp + 16);
    return f.h;
}

DEV v8f wmma_f16(v16h a, v16h b, v8f c) {
    return __builtin_amdgcn_wmma_f32_16x16x32_f16(false, a, false, b, (short)0, c,
                                                  false, false);
}

// CDNA5 async copy: 16 bytes global -> LDS per lane, tracked by ASYNCcnt.
DEV void async_g2l_b128(const _Float16* gptr, _Float16* lptr) {
    unsigned lds_off = (unsigned)(unsigned long long)(lds_f16_t*)lptr;
    asm volatile("global_load_async_to_lds_b128 %0, %1, off"
                 :: "v"(lds_off), "v"(gptr) : "memory");
}
DEV void async_wait0() {
    asm volatile("s_wait_asynccnt 0x0" ::: "memory");
}

// ---------------------------------------------------------------------------
// K0: transpose + convert f32 W[K][N] -> f16 Wt[N][K]
// ---------------------------------------------------------------------------
__global__ void k_pack_t(const float* __restrict__ W, _Float16* __restrict__ Wt,
                         int K, int N) {
    int idx = blockIdx.x * blockDim.x + threadIdx.x;
    if (idx >= K * N) return;
    int k = idx / N, n = idx % N;
    Wt[n * K + k] = (_Float16)W[idx];
}

// ---------------------------------------------------------------------------
// K1: LayerNorm over last dim (512), output f16.  One block (256 thr) per row.
// ---------------------------------------------------------------------------
__global__ void k_layernorm(const float* __restrict__ x, const float* __restrict__ gamma,
                            _Float16* __restrict__ xn) {
    int row = blockIdx.x;
    int t   = threadIdx.x;
    const float* xr = x + (size_t)row * DIMK;
    __shared__ float red[8];
    __shared__ float stat[2];

    float v0 = xr[t], v1 = xr[t + 256];
    float s = v0 + v1;
    for (int m = 16; m > 0; m >>= 1) s += __shfl_xor(s, m, 32);
    if ((t & 31) == 0) red[t >> 5] = s;
    __syncthreads();
    if (t == 0) { float a = 0.f; for (int i = 0; i < 8; i++) a += red[i]; stat[0] = a * (1.0f / DIMK); }
    __syncthreads();
    float mean = stat[0];
    float d0 = v0 - mean, d1 = v1 - mean;
    s = d0 * d0 + d1 * d1;
    for (int m = 16; m > 0; m >>= 1) s += __shfl_xor(s, m, 32);
    __syncthreads();
    if ((t & 31) == 0) red[t >> 5] = s;
    __syncthreads();
    if (t == 0) { float a = 0.f; for (int i = 0; i < 8; i++) a += red[i]; stat[1] = rsqrtf(a * (1.0f / DIMK) + 1e-5f); }
    __syncthreads();
    float rs = stat[1];
    xn[(size_t)row * DIMK + t]       = (_Float16)(d0 * rs * gamma[t]);
    xn[(size_t)row * DIMK + t + 256] = (_Float16)(d1 * rs * gamma[t + 256]);
}

// ---------------------------------------------------------------------------
// K2: register-blocked WMMA GEMM: C[M][N] (f32) = A[M][K] (f16) @ Bt[N][K]^T.
// Block = 256 thr = 8 waves (2x4).  Wave tile = 32x64 (2 A-frags x 4 B-frags,
// 8 accumulators -> 8 WMMAs per 6 fragment loads per K-step).
// Block tile = 64x256.  grid = (M/64, N/256)
// ---------------------------------------------------------------------------
__global__ void k_gemm_f16(const _Float16* __restrict__ A, const _Float16* __restrict__ Bt,
                           float* __restrict__ C, int M, int N, int K) {
    int wave = threadIdx.x >> 5, lane = threadIdx.x & 31;
    int row0 = blockIdx.x * 64 + (wave >> 2) * 32;
    int col0 = blockIdx.y * 256 + (wave & 3) * 64;
    if (row0 >= M) return;

    const _Float16* a_base = A  + (size_t)row0 * K;
    const _Float16* b_base = Bt + (size_t)col0 * K;
    v8f acc[2][4] = {};
    for (int k0 = 0; k0 < K; k0 += 32) {
        v16h af0 = load_a_frag(a_base + k0, K, lane);
        v16h af1 = load_a_frag(a_base + (size_t)16 * K + k0, K, lane);
        #pragma unroll
        for (int j = 0; j < 4; j++) {
            v16h bf = load_bt_frag(b_base + (size_t)(j * 16) * K + k0, K, lane);
            acc[0][j] = wmma_f16(af0, bf, acc[0][j]);
            acc[1][j] = wmma_f16(af1, bf, acc[1][j]);
        }
    }
    int c  = lane & 15;
    int rb = (lane >> 4) * 8;
    #pragma unroll
    for (int i = 0; i < 2; i++)
        #pragma unroll
        for (int j = 0; j < 4; j++) {
            float* out = C + (size_t)(row0 + i * 16 + rb) * N + col0 + j * 16 + c;
            #pragma unroll
            for (int t = 0; t < 8; t++) out[(size_t)t * N] = acc[i][j][t];
        }
}

// ---------------------------------------------------------------------------
// K3: per-head L2 norm + scale for q/k, reorg v to transposed layout.
// ---------------------------------------------------------------------------
__global__ void k_qkv_post(const float* __restrict__ qkv,
                           const float* __restrict__ q_scale, const float* __restrict__ k_scale,
                           _Float16* __restrict__ qh, _Float16* __restrict__ kh,
                           _Float16* __restrict__ vT) {
    int gw   = blockIdx.x * 8 + (threadIdx.x >> 5);
    int lane = threadIdx.x & 31;
    int which = gw % 3;
    int t     = gw / 3;              // [0, MROWS*HEADS)
    int h   = t % HEADS;
    int row = t / HEADS;             // b*NN + n
    int b = row / NN, n = row % NN;
    int d0 = lane * 2, d1 = d0 + 1;

    const float* src = qkv + (size_t)row * (3 * INNER) + which * INNER + h * DH;
    float x0 = src[d0], x1 = src[d1];

    if (which < 2) {
        float ss = x0 * x0 + x1 * x1;
        for (int m = 16; m > 0; m >>= 1) ss += __shfl_xor(ss, m, 32);
        float inv = 1.0f / fmaxf(sqrtf(ss), 1e-12f);
        const float* sc = (which == 0) ? q_scale : k_scale;
        x0 = x0 * inv * sc[d0];
        x1 = x1 * inv * sc[d1];
        _Float16* dst = ((which == 0) ? qh : kh) + ((size_t)(b * HEADS + h) * NN + n) * DH + d0;
        dst[0] = (_Float16)x0;
        dst[1] = (_Float16)x1;
    } else {
        _Float16* dst = vT + (size_t)(b * HEADS + h) * DH * NN + n;
        dst[(size_t)d0 * NN] = (_Float16)x0;
        dst[(size_t)d1 * NN] = (_Float16)x1;
    }
}

// ---------------------------------------------------------------------------
// K4: fused CPB MLP.  Per block: 16 positions.
//   h0 = leaky(rel @ w0 + b0)                         (scalar -> LDS f16)
//   h1 = leaky(h0 @ w1 + b1)   WMMA; w1T tile staged to LDS with CDNA5
//                              global_load_async_to_lds_b128 (ASYNCcnt)
//   bias = h1 @ w2 + b2        (scalar 512 -> 8)
// ---------------------------------------------------------------------------
__global__ void k_cpb(const float* __restrict__ w0, const float* __restrict__ b0,
                      const _Float16* __restrict__ w1T, const float* __restrict__ b1,
                      const float* __restrict__ w2, const float* __restrict__ b2,
                      float* __restrict__ bias) {
    __shared__ _Float16 h0[16][DIMK];       // 16 KB
    __shared__ _Float16 h1[16][DIMK];       // 16 KB
    __shared__ _Float16 btile[256][32];     // 16 KB (256 cols x 32 K slice of w1T)
    int p0 = blockIdx.x * 16;
    int wave = threadIdx.x >> 5, lane = threadIdx.x & 31;

    // stage 1: rel -> h0 (16 x 512)
    for (int e = threadIdx.x; e < 16 * DIMK; e += 256) {
        int r = e >> 9, c = e & 511;
        int p = p0 + r;
        int i = p / NN, j = p % NN;
        int di = (i / 24) - (j / 24);
        int dj = (i % 24) - (j % 24);
        float r0 = copysignf(log1pf(fabsf((float)di)), (float)di);
        float r1 = copysignf(log1pf(fabsf((float)dj)), (float)dj);
        float a = r0 * w0[c] + r1 * w0[DIMK + c] + b0[c];
        h0[r][c] = (_Float16)((a > 0.f) ? a : LEAK * a);
    }
    __syncthreads();

    // stage 2: h0 @ w1, two 256-col halves; per K-step stage B slice via async DMA
    for (int half = 0; half < 2; half++) {
        v8f acc2[2] = {};
        for (int k0 = 0; k0 < DIMK; k0 += 32) {
            // async stage: each thread copies its 64-byte row chunk (4 x b128)
            const _Float16* grow = w1T + (size_t)(half * 256 + threadIdx.x) * DIMK + k0;
            _Float16*       lrow = &btile[threadIdx.x][0];
            #pragma unroll
            for (int it = 0; it < 4; it++)
                async_g2l_b128(grow + it * 8, lrow + it * 8);
            async_wait0();
            __syncthreads();

            v16h a = load_a_frag(&h0[0][0] + k0, DIMK, lane);
            #pragma unroll
            for (int j = 0; j < 2; j++) {
                v16h bf = load_bt_frag(&btile[(wave * 2 + j) * 16][0], 32, lane);
                acc2[j] = wmma_f16(a, bf, acc2[j]);
            }
            __syncthreads();   // btile reuse fence
        }
        // epilogue -> h1 (leaky(acc + b1))
        int c  = lane & 15;
        int rb = (lane >> 4) * 8;
        #pragma unroll
        for (int j = 0; j < 2; j++) {
            int col0 = half * 256 + (wave * 2 + j) * 16;
            for (int jj = 0; jj < 8; jj++) {
                float v = acc2[j][jj] + b1[col0 + c];
                h1[rb + jj][col0 + c] = (_Float16)((v > 0.f) ? v : LEAK * v);
            }
        }
    }
    __syncthreads();

    // stage 3: h1 @ w2 (512 -> 8 heads)
    if (threadIdx.x < 128) {
        int r = threadIdx.x >> 3, o = threadIdx.x & 7;
        float s = b2[o];
        for (int k = 0; k < DIMK; k++) s += (float)h1[r][k] * w2[k * HEADS + o];
        int p = p0 + r;
        int i = p / NN, j = p % NN;
        bias[((size_t)o * NN + i) * NN + j] = s;
    }
}

// ---------------------------------------------------------------------------
// K5: attention.  grid = (NN/16, B*H).  Block = 256 thr = 8 waves.
// ---------------------------------------------------------------------------
#define SLD 592   // padded LDS leading dim (multiple of 8 halves)

__global__ void k_attn(const _Float16* __restrict__ qh, const _Float16* __restrict__ kh,
                       const _Float16* __restrict__ vT, const float* __restrict__ bias,
                       _Float16* __restrict__ attn_out) {
    __shared__ float    s[16][SLD];
    __shared__ _Float16 p[16][SLD];
    __shared__ float    redm[16][17];
    __shared__ float    inv[16];

    int bh = blockIdx.y;
    int h  = bh & (HEADS - 1);
    int b  = bh >> 3;
    int i0 = blockIdx.x * 16;
    int wave = threadIdx.x >> 5, lane = threadIdx.x & 31;

    // ---- sim = q @ k^T * SCALE + bias (q fragments hoisted: invariant in j)
    const _Float16* qbase = qh + ((size_t)bh * NN + i0) * DH;
    v16h qa0 = load_a_frag(qbase,      DH, lane);
    v16h qa1 = load_a_frag(qbase + 32, DH, lane);
    for (int jt = wave; jt < NN / 16; jt += 8) {
        int j0 = jt * 16;
        const _Float16* kbase = kh + ((size_t)bh * NN + j0) * DH;
        v8f acc = {};
        acc = wmma_f16(qa0, load_bt_frag(kbase,      DH, lane), acc);
        acc = wmma_f16(qa1, load_bt_frag(kbase + 32, DH, lane), acc);
        int c  = lane & 15;
        int rb = (lane >> 4) * 8;
        const float* bp = bias + ((size_t)(h * NN + i0 + rb)) * NN + j0 + c;
        for (int jj = 0; jj < 8; jj++)
            s[rb + jj][j0 + c] = acc[jj] * SCALE_F + bp[(size_t)jj * NN];
    }
    __syncthreads();

    // ---- softmax (unnormalized exp -> p, 1/sum -> inv)
    {
        int r = threadIdx.x & 15, g = threadIdx.x >> 4;   // 16 groups x 36 cols
        float m = -1e30f;
        for (int c = g * 36; c < g * 36 + 36; c++) m = fmaxf(m, s[r][c]);
        redm[r][g] = m;
        __syncthreads();
        if (g == 0) {
            float mm = -1e30f;
            for (int q2 = 0; q2 < 16; q2++) mm = fmaxf(mm, redm[r][q2]);
            redm[r][16] = mm;
        }
        __syncthreads();
        float rm  = redm[r][16];
        float sum = 0.f;
        for (int c = g * 36; c < g * 36 + 36; c++) {
            float e = __expf(s[r][c] - rm);
            p[r][c] = (_Float16)e;
            sum += e;
        }
        __syncthreads();
        redm[r][g] = sum;
        __syncthreads();
        if (g == 0) {
            float t = 0.f;
            for (int q2 = 0; q2 < 16; q2++) t += redm[r][q2];
            inv[r] = 1.0f / t;
        }
        __syncthreads();
    }

    // ---- out = (p @ v) * inv
    if (wave < 4) {
        int n0 = wave * 16;   // d-dim tile
        const _Float16* vb = vT + (size_t)bh * DH * NN + (size_t)n0 * NN;
        v8f acc = {};
        for (int k0 = 0; k0 < NN; k0 += 32) {
            v16h a   = load_a_frag(&p[0][0] + k0, SLD, lane);
            v16h bfr = load_bt_frag(vb + k0, NN, lane);
            acc = wmma_f16(a, bfr, acc);
        }
        int c  = lane & 15;
        int rb = (lane >> 4) * 8;
        for (int jj = 0; jj < 8; jj++) {
            float v = acc[jj] * inv[rb + jj];
            attn_out[((size_t)(b * NN + i0 + rb + jj)) * INNER + h * DH + n0 + c] = (_Float16)v;
        }
    }
}

// ---------------------------------------------------------------------------
// launch
// ---------------------------------------------------------------------------
extern "C" void kernel_launch(void* const* d_in, const int* in_sizes, int n_in,
                              void* d_out, int out_size, void* d_ws, size_t ws_size,
                              hipStream_t stream) {
    (void)in_sizes; (void)n_in; (void)out_size; (void)ws_size;
    const float* x       = (const float*)d_in[0];
    const float* gamma   = (const float*)d_in[1];
    const float* Wq      = (const float*)d_in[2];
    const float* Wkv     = (const float*)d_in[3];
    const float* Wo      = (const float*)d_in[4];
    const float* q_scale = (const float*)d_in[5];
    const float* k_scale = (const float*)d_in[6];
    const float* cpb_w0  = (const float*)d_in[7];
    const float* cpb_b0  = (const float*)d_in[8];
    const float* cpb_w1  = (const float*)d_in[9];
    const float* cpb_b1  = (const float*)d_in[10];
    const float* cpb_w2  = (const float*)d_in[11];
    const float* cpb_b2  = (const float*)d_in[12];

    uintptr_t w = ((uintptr_t)d_ws + 255) & ~(uintptr_t)255;
    auto carve = [&](size_t bytes) -> uintptr_t {
        uintptr_t pp = w;
        w = (w + bytes + 255) & ~(uintptr_t)255;
        return pp;
    };

    _Float16* xn     = (_Float16*)carve((size_t)MROWS * DIMK * 2);
    _Float16* WqkvT  = (_Float16*)carve((size_t)(3 * INNER) * DIMK * 2);
    _Float16* WoT    = (_Float16*)carve((size_t)DIMK * INNER * 2);
    _Float16* W1T    = (_Float16*)carve((size_t)DIMK * DIMK * 2);
    float*    qkv    = (float*)   carve((size_t)MROWS * (3 * INNER) * 4);
    _Float16* qhd    = (_Float16*)carve((size_t)BB * HEADS * NN * DH * 2);
    _Float16* khd    = (_Float16*)carve((size_t)BB * HEADS * NN * DH * 2);
    _Float16* vTd    = (_Float16*)carve((size_t)BB * HEADS * DH * NN * 2);
    _Float16* attn_o = (_Float16*)carve((size_t)MROWS * INNER * 2);
    float*    biasb  = (float*)   carve((size_t)HEADS * NN * NN * 4);

    // weight packing (transpose + f32->f16)
    k_pack_t<<<(DIMK * INNER + 255) / 256, 256, 0, stream>>>(Wq, WqkvT, DIMK, INNER);
    k_pack_t<<<(DIMK * 2 * INNER + 255) / 256, 256, 0, stream>>>(Wkv, WqkvT + (size_t)INNER * DIMK, DIMK, 2 * INNER);
    k_pack_t<<<(INNER * DIMK + 255) / 256, 256, 0, stream>>>(Wo, WoT, INNER, DIMK);
    k_pack_t<<<(DIMK * DIMK + 255) / 256, 256, 0, stream>>>(cpb_w1, W1T, DIMK, DIMK);

    // layernorm
    k_layernorm<<<MROWS, 256, 0, stream>>>(x, gamma, xn);

    // qkv projection: [9216, 512] @ [512, 1536]
    k_gemm_f16<<<dim3(MROWS / 64, (3 * INNER) / 256), 256, 0, stream>>>(
        xn, WqkvT, qkv, MROWS, 3 * INNER, DIMK);

    // l2norm + reorg
    k_qkv_post<<<(MROWS * HEADS * 3) / 8, 256, 0, stream>>>(
        qkv, q_scale, k_scale, qhd, khd, vTd);

    // fused CPB bias MLP (dominant FLOPs)
    k_cpb<<<PPOS / 16, 256, 0, stream>>>(cpb_w0, cpb_b0, W1T, cpb_b1, cpb_w2, cpb_b2, biasb);

    // attention
    k_attn<<<dim3(NN / 16, BB * HEADS), 256, 0, stream>>>(qhd, khd, vTd, biasb, attn_o);

    // output projection -> d_out (f32)
    k_gemm_f16<<<dim3(MROWS / 64, DIMK / 256), 256, 0, stream>>>(
        attn_o, WoT, (float*)d_out, MROWS, DIMK, DIMK);
}
